// TempGAT_70695161692664
// MI455X (gfx1250) — matrix operand order repaired and checked by reading, not compiled, not run on hardware
//
#include <hip/hip_runtime.h>

typedef __attribute__((ext_vector_type(16))) __bf16 v16bf;
typedef __attribute__((ext_vector_type(8)))  __bf16 v8bf;
typedef __attribute__((ext_vector_type(8)))  float  v8f;

#define N_NODES 4096
#define FIN     256
#define NHEAD   8
#define FH      64
#define KOUT    512
#define NEGV    (-9e15f)
#define LALPHA  0.2f

// round-to-nearest-even f32 -> bf16
__device__ __forceinline__ __bf16 f2bf(float f) {
    union { float f; unsigned u; } c; c.f = f;
    unsigned r = c.u + 0x7FFFu + ((c.u >> 16) & 1u);
    union { unsigned short s; __bf16 b; } o;
    o.s = (unsigned short)(r >> 16);
    return o.b;
}

// Packed B-fragment index for a [K x 64] bf16 matrix consumed as WMMA B:
// element (k, col) of fragment block kb=k&~31, nt=col>>4 is stored so that
// lane (half*16+m16) reads its 16 K-elements contiguously (32B per lane).
__device__ __forceinline__ long pk_idx(int k, int col) {
    int kb   = k >> 5;          // 32-row K-block
    int half = (k >> 3) & 1;
    int e    = (k & 7) | (((k >> 4) & 1) << 3);
    int nt   = col >> 4;
    int m16  = col & 15;
    return ((long)(kb * 4 + nt) << 9) + ((half * 16 + m16) << 4) + e;
}

__global__ void cvt_f32_bf16(const float* __restrict__ src,
                             __bf16* __restrict__ dst, int n) {
    int i = blockIdx.x * blockDim.x + threadIdx.x;
    if (i < n) dst[i] = f2bf(src[i]);
}

// Pack a row-major f32 [K x 64] weight matrix into bf16 B-fragment order.
// One thread per packed element; batched over blockIdx.y.
__global__ void pack_b_frags(const float* __restrict__ src,
                             __bf16* __restrict__ dst, int K) {
    const int total = K * 64;
    int o = blockIdx.x * blockDim.x + threadIdx.x;
    if (o >= total) return;
    const int b = blockIdx.y;
    src += (long)b * total;
    dst += (long)b * total;
    int e    = o & 15;
    int lane = (o >> 4) & 31;
    int frag = o >> 9;
    int nt   = frag & 3;
    int kb   = (frag >> 2) << 5;
    int half = lane >> 4, m16 = lane & 15;
    int k = kb + ((e >> 3) << 4) + half * 8 + (e & 7);
    dst[o] = f2bf(src[(long)k * 64 + nt * 16 + m16]);
}

// C[b] = A[b] @ B[b]; A:[M x K] bf16 row-major, Bp: packed B fragments.
// Block = 256 threads = 8 waves; each wave owns a 16x64 output tile.
// Writes Cf (row-major f32) and Cb (packed B-fragment bf16, ready to be the
// B operand of the downstream attention WMMA).
__global__ void gemm_bf16_n64(const __bf16* __restrict__ A,
                              const __bf16* __restrict__ Bp,
                              float* __restrict__ Cf, __bf16* __restrict__ Cb,
                              int K, long aBatchStride, long bBatchStride,
                              long cBatchStride) {
    const int lane = threadIdx.x & 31;
    const int wave = threadIdx.x >> 5;
    const int half = (lane >> 4) & 1;
    const int m16  = lane & 15;
    const int row0 = blockIdx.x * 128 + wave * 16;
    const int b    = blockIdx.y;

    const __bf16* Ab = A  + (long)b * aBatchStride;
    const __bf16* Bb = Bp + (long)b * bBatchStride;

    v8f acc[4] = {};
    for (int kb = 0; kb < K; kb += 32) {
        // A fragment 16x32: lane m16 holds K = half*8 + {0..7} and 16+half*8+{0..7}
        const __bf16* ap = Ab + (long)(row0 + m16) * K + kb + half * 8;
        v8bf lo = *(const v8bf*)(ap);
        v8bf hi = *(const v8bf*)(ap + 16);
        v16bf af = __builtin_shufflevector(lo, hi, 0,1,2,3,4,5,6,7,
                                           8,9,10,11,12,13,14,15);
        const __bf16* bp = Bb + (((long)(kb >> 5) * 4) << 9) + lane * 16;
        #pragma unroll
        for (int nt = 0; nt < 4; ++nt) {
            v16bf bf = *(const v16bf*)(bp + ((long)nt << 9));
            acc[nt] = __builtin_amdgcn_wmma_f32_16x16x32_bf16(
                false, af, false, bf, (short)0, acc[nt], false, false);
        }
    }
    float*  Cfb = Cf + (long)b * cBatchStride;
    __bf16* Cbb = Cb + (long)b * cBatchStride;
    #pragma unroll
    for (int nt = 0; nt < 4; ++nt)
        #pragma unroll
        for (int r = 0; r < 8; ++r) {
            int row = row0 + r + half * 8;   // C/D layout: VGPR r -> M = r + 8*half
            int col = nt * 16 + m16;
            float v = acc[nt][r];
            Cfb[(long)row * 64 + col] = v;
            Cbb[pk_idx(row, col)] = f2bf(v);
        }
}

// f1[i] = h[i,:] . a[0:64],  f2[i] = h[i,:] . a[64:128], per head
__global__ void fvec_kernel(const float* __restrict__ Hm,
                            const float* __restrict__ a,
                            float* __restrict__ f1, float* __restrict__ f2,
                            int rowsTotal) {
    int tid = blockIdx.x * blockDim.x + threadIdx.x;
    if (tid >= rowsTotal) return;
    int head = tid >> 12;                 // 4096 rows per head
    const float* hr = Hm + (long)tid * 64;
    const float* ah = a + (long)head * 128;
    float s1 = 0.f, s2 = 0.f;
    #pragma unroll 8
    for (int f = 0; f < 64; ++f) {
        float hv = hr[f];
        s1 += hv * ah[f];
        s2 += hv * ah[64 + f];
    }
    f1[tid] = s1; f2[tid] = s2;
}

// Per-row softmax statistics: rowmax and rowsum of exp(e - rowmax).
// One block (256 threads) per (row, head); adj row is L2-resident.
__global__ void rowstat_kernel(const float* __restrict__ f1,
                               const float* __restrict__ f2,
                               const float* __restrict__ adj,
                               float* __restrict__ rowmax,
                               float* __restrict__ rowsum) {
    const int row = blockIdx.x, head = blockIdx.y, t = threadIdx.x;
    const float  f1r  = f1[head * N_NODES + row];
    const float* f2h  = f2 + head * N_NODES;
    const float* arow = adj + (long)row * N_NODES;

    float ev[16];
    float mymax = -INFINITY;
    #pragma unroll
    for (int s = 0; s < 16; ++s) {
        int j = s * 256 + t;
        float sc = f1r + f2h[j];
        sc = sc > 0.f ? sc : LALPHA * sc;
        sc = (arow[j] > 0.f) ? sc : NEGV;
        ev[s] = sc;
        mymax = fmaxf(mymax, sc);
    }
    __shared__ float red[256];
    red[t] = mymax; __syncthreads();
    for (int o = 128; o > 0; o >>= 1) {
        if (t < o) red[t] = fmaxf(red[t], red[t + o]);
        __syncthreads();
    }
    float rmax = red[0]; __syncthreads();
    float mysum = 0.f;
    #pragma unroll
    for (int s = 0; s < 16; ++s) mysum += __expf(ev[s] - rmax);
    red[t] = mysum; __syncthreads();
    for (int o = 128; o > 0; o >>= 1) {
        if (t < o) red[t] += red[t + o];
        __syncthreads();
    }
    if (t == 0) {
        rowmax[head * N_NODES + row] = rmax;
        rowsum[head * N_NODES + row] = red[0];
    }
}

// Flash-style fused attention: out = softmax(mask(leakyrelu(f1_i+f2_j))) @ H.
// Block = 8 waves handling one 16-row tile; waves split the j dimension,
// build P directly in the WMMA A-fragment layout (score inputs come in as
// aligned float8 vector loads), pull H as packed B fragments (32B/lane), and
// accumulate with v_wmma_f32_16x16x32_bf16. Partial D summed through LDS.
__global__ void attn_kernel(const float* __restrict__ f1,
                            const float* __restrict__ f2,
                            const float* __restrict__ adj,
                            const __bf16* __restrict__ Hp,   // packed fragments
                            const float* __restrict__ rowmax,
                            const float* __restrict__ rowsum,
                            __bf16* __restrict__ outBf, float* __restrict__ outF,
                            int outLd, int applyElu) {
    const int lane = threadIdx.x & 31;
    const int wave = threadIdx.x >> 5;
    const int half = (lane >> 4) & 1;
    const int m16  = lane & 15;
    const int head = blockIdx.y;
    const int i0   = blockIdx.x * 16;

    const float*  f2h = f2 + head * N_NODES;
    const __bf16* Hh  = Hp + (long)head * N_NODES * 64;
    const int myrow   = i0 + m16;                    // A-fragment row for this lane
    const float f1r   = f1[head * N_NODES + myrow];
    const float rmax  = rowmax[head * N_NODES + myrow];
    const float* arow = adj + (long)myrow * N_NODES;

    v8f acc[4] = {};
    for (int j0 = wave * 32; j0 < N_NODES; j0 += 256) {
        // lane's 16 K-elements are two contiguous 8-float runs
        const float* fp = f2h  + j0 + half * 8;
        const float* mp = arow + j0 + half * 8;
        v8f f2lo = *(const v8f*)(fp);
        v8f f2hi = *(const v8f*)(fp + 16);
        v8f adlo = *(const v8f*)(mp);
        v8f adhi = *(const v8f*)(mp + 16);
        v16bf af;
        #pragma unroll
        for (int e = 0; e < 8; ++e) {
            float sc = f1r + f2lo[e];
            sc = sc > 0.f ? sc : LALPHA * sc;
            sc = (adlo[e] > 0.f) ? sc : NEGV;
            af[e] = f2bf(__expf(sc - rmax));
            float sh = f1r + f2hi[e];
            sh = sh > 0.f ? sh : LALPHA * sh;
            sh = (adhi[e] > 0.f) ? sh : NEGV;
            af[e + 8] = f2bf(__expf(sh - rmax));
        }
        const __bf16* bp = Hh + (((long)(j0 >> 5) * 4) << 9) + lane * 16;
        #pragma unroll
        for (int nt = 0; nt < 4; ++nt) {
            v16bf bf = *(const v16bf*)(bp + ((long)nt << 9));
            acc[nt] = __builtin_amdgcn_wmma_f32_16x16x32_bf16(
                false, af, false, bf, (short)0, acc[nt], false, false);
        }
    }

    // cross-wave reduction of the 16x64 partial output tile
    __shared__ float sD[16 * 64];
    for (int idx = threadIdx.x; idx < 1024; idx += 256) sD[idx] = 0.f;
    __syncthreads();
    #pragma unroll
    for (int nt = 0; nt < 4; ++nt)
        #pragma unroll
        for (int r = 0; r < 8; ++r)
            atomicAdd(&sD[(r + half * 8) * 64 + nt * 16 + m16], acc[nt][r]);
    __syncthreads();

    for (int idx = threadIdx.x; idx < 1024; idx += 256) {
        int m = idx >> 6, n = idx & 63;
        int row = i0 + m;
        float v = sD[idx] / rowsum[head * N_NODES + row];
        if (applyElu) v = v > 0.f ? v : (__expf(v) - 1.f);
        if (outBf) outBf[(long)row * outLd + head * 64 + n] = f2bf(v);
        else       outF [(long)row * outLd + n] = v;
    }
}

extern "C" void kernel_launch(void* const* d_in, const int* in_sizes, int n_in,
                              void* d_out, int out_size, void* d_ws, size_t ws_size,
                              hipStream_t stream) {
    const float* features = (const float*)d_in[0];   // [4096,256]
    const float* adj      = (const float*)d_in[1];   // [4096,4096]
    const float* W_heads  = (const float*)d_in[2];   // [8,256,64]
    const float* a_heads  = (const float*)d_in[3];   // [8,128,1]
    const float* W_out    = (const float*)d_in[4];   // [512,64]
    const float* a_out    = (const float*)d_in[5];   // [128,1]
    float* out = (float*)d_out;                      // [4096,64]

    char* ws = (char*)d_ws;
    size_t off = 0;
    auto alloc = [&](size_t bytes) -> void* {
        void* p = ws + off;
        off += (bytes + 255) & ~(size_t)255;
        return p;
    };

    __bf16* featB = (__bf16*)alloc((size_t)N_NODES * FIN * 2);
    __bf16* WhP   = (__bf16*)alloc((size_t)NHEAD * FIN * FH * 2);   // packed frags
    __bf16* WoP   = (__bf16*)alloc((size_t)KOUT * FH * 2);          // packed frags
    float*  hF    = (float*) alloc((size_t)NHEAD * N_NODES * FH * 4);
    __bf16* hP    = (__bf16*)alloc((size_t)NHEAD * N_NODES * FH * 2); // packed frags
    float*  f1    = (float*) alloc((size_t)NHEAD * N_NODES * 4);
    float*  f2    = (float*) alloc((size_t)NHEAD * N_NODES * 4);
    float*  rmax  = (float*) alloc((size_t)NHEAD * N_NODES * 4);
    float*  rsum  = (float*) alloc((size_t)NHEAD * N_NODES * 4);
    __bf16* x2B   = (__bf16*)alloc((size_t)N_NODES * KOUT * 2);     // row-major (A op)
    float*  h2F   = (float*) alloc((size_t)N_NODES * FH * 4);
    __bf16* h2P   = (__bf16*)alloc((size_t)N_NODES * FH * 2);       // packed frags
    float*  f1o   = (float*) alloc((size_t)N_NODES * 4);
    float*  f2o   = (float*) alloc((size_t)N_NODES * 4);
    float*  rmaxo = (float*) alloc((size_t)N_NODES * 4);
    float*  rsumo = (float*) alloc((size_t)N_NODES * 4);

    // 1) operand preparation: features -> bf16 row-major; weights -> packed frags
    {
        int n = N_NODES * FIN;
        cvt_f32_bf16<<<(n + 255) / 256, 256, 0, stream>>>(features, featB, n);
        pack_b_frags<<<dim3((FIN * FH + 255) / 256, NHEAD), 256, 0, stream>>>(
            W_heads, WhP, FIN);
        pack_b_frags<<<dim3((KOUT * FH + 255) / 256, 1), 256, 0, stream>>>(
            W_out, WoP, KOUT);
    }

    // 2) per-head h = x @ W   (batched over 8 heads; A shared)
    gemm_bf16_n64<<<dim3(N_NODES / 128, NHEAD), 256, 0, stream>>>(
        featB, WhP, hF, hP, FIN,
        /*aBatch*/0L, (long)FIN * FH, (long)N_NODES * FH);

    // 3) f1/f2 score vectors per head
    fvec_kernel<<<(NHEAD * N_NODES + 255) / 256, 256, 0, stream>>>(
        hF, a_heads, f1, f2, NHEAD * N_NODES);

    // 4) softmax row statistics
    rowstat_kernel<<<dim3(N_NODES, NHEAD), 256, 0, stream>>>(f1, f2, adj, rmax, rsum);

    // 5) fused attention + ELU -> concatenated bf16 features [4096, 512]
    attn_kernel<<<dim3(N_NODES / 16, NHEAD), 256, 0, stream>>>(
        f1, f2, adj, hP, rmax, rsum, x2B, nullptr, KOUT, /*elu=*/1);

    // 6) output layer: h2 = x2 @ W_out
    gemm_bf16_n64<<<dim3(N_NODES / 128, 1), 256, 0, stream>>>(
        x2B, WoP, h2F, h2P, KOUT, 0L, 0L, 0L);

    // 7) output-layer score vectors
    fvec_kernel<<<(N_NODES + 255) / 256, 256, 0, stream>>>(
        h2F, a_out, f1o, f2o, N_NODES);

    // 8) output-layer softmax stats
    rowstat_kernel<<<dim3(N_NODES, 1), 256, 0, stream>>>(f1o, f2o, adj, rmaxo, rsumo);

    // 9) final fused attention -> fp32 output (no ELU)
    attn_kernel<<<dim3(N_NODES / 16, 1), 256, 0, stream>>>(
        f1o, f2o, adj, h2P, rmaxo, rsumo, nullptr, out, FH, /*elu=*/0);
}
